// PC_RNN_V_14413910245616
// MI455X (gfx1250) — matrix operand order repaired and checked by reading, not compiled, not run on hardware
//
#include <hip/hip_runtime.h>
#include <stdint.h>

typedef __attribute__((ext_vector_type(16))) __bf16 v16bf;
typedef __attribute__((ext_vector_type(8)))  float  v8f;

#define SEQ_LEN 1024
#define BATCH   128
#define SDIM    512
#define ODIM    64
#define TAU_INV 0.5f
#define ALPHA_X 0.1f

__device__ __forceinline__ __bf16 f32_to_bf16(float f) {
  union { float f; uint32_t u; } in; in.f = f;
  uint32_t u = in.u;
  uint32_t r = (u + 0x7FFFu + ((u >> 16) & 1u)) >> 16;   // round-to-nearest-even
  union { uint16_t u; __bf16 b; } out; out.u = (uint16_t)r;
  return out.b;
}

union FragAB { uint4 q[2]; v16bf v; };
union FragC  { v8f v; float f[8]; };

// ---------------------------------------------------------------------------
// Prep: fp32 weights -> bf16 (w_r, w_o row-major, w_o transposed) in scratch.
// ---------------------------------------------------------------------------
__global__ void pc_rnn_prep_kernel(const float* __restrict__ wr,
                                   const float* __restrict__ wo,
                                   __bf16* __restrict__ wr_b,
                                   __bf16* __restrict__ wo_b,
                                   __bf16* __restrict__ woT_b) {
  int i = blockIdx.x * blockDim.x + threadIdx.x;
  if (i < SDIM * SDIM) wr_b[i] = f32_to_bf16(wr[i]);
  if (i < ODIM * SDIM) {
    __bf16 v = f32_to_bf16(wo[i]);
    wo_b[i] = v;
    int o = i / SDIM, s = i - o * SDIM;
    woT_b[s * ODIM + o] = v;
  }
}

// ---------------------------------------------------------------------------
// Main: one workgroup per 16 batch rows (recurrence independent across batch).
// 256 threads = 8 waves; each wave owns 64 state columns (4 N-tiles of 16).
// All matmuls via v_wmma_f32_16x16x32_bf16.
// ---------------------------------------------------------------------------
__launch_bounds__(256, 1)
__global__ void pc_rnn_main_kernel(const float*  __restrict__ x,
                                   const float*  __restrict__ h_init,
                                   const float*  __restrict__ b_o,
                                   const float*  __restrict__ b_r,
                                   const __bf16* __restrict__ wr_b,   // [512][512]
                                   const __bf16* __restrict__ wo_b,   // [64][512]
                                   const __bf16* __restrict__ woT_b,  // [512][64]
                                   float* __restrict__ out)           // [1024][128][64]
{
  __shared__ __bf16 lds_th[16 * SDIM];     // tanh(h_post)  16KB
  __shared__ __bf16 lds_tp[16 * SDIM];     // tanh(h_prior) 16KB
  __shared__ __bf16 lds_er[16 * ODIM];     // error          2KB
  __shared__ __bf16 lds_woT[SDIM * ODIM];  // w_o^T bf16    64KB
  __shared__ float  lds_br[SDIM];
  __shared__ float  lds_bo[ODIM];

  const int tid  = threadIdx.x;
  const int wave = tid >> 5;       // 0..7
  const int lane = tid & 31;
  const int lo16 = lane & 15;
  const int hi   = lane >> 4;      // 0 or 1
  const int m0   = blockIdx.x * 16;        // batch-row base for this WG
  const int nbase = wave * 64;             // state-column base for this wave

  // Stage constants into LDS (uint4 copies, fully coalesced).
  for (int i = tid; i < (SDIM * ODIM) / 8; i += 256)
    ((uint4*)lds_woT)[i] = ((const uint4*)woT_b)[i];
  for (int i = tid; i < SDIM; i += 256) lds_br[i] = b_r[i];
  if (tid < ODIM) lds_bo[tid] = b_o[tid];

  // Per-wave recurrent state: 4 accumulator tiles (C/D layout: n=lo16, m=r+hi*8).
  FragC h[4], tp[4];
  #pragma unroll
  for (int t4 = 0; t4 < 4; ++t4) {
    int col = nbase + t4 * 16 + lo16;
    #pragma unroll
    for (int r = 0; r < 8; ++r)
      h[t4].f[r] = h_init[(size_t)(m0 + r + hi * 8) * SDIM + col];
  }
  __syncthreads();

  for (int t = 0; t < SEQ_LEN; ++t) {
    // ---- phase A: th = tanh(h_post) -> LDS (bf16, row-major [16][512])
    #pragma unroll
    for (int t4 = 0; t4 < 4; ++t4) {
      int col = nbase + t4 * 16 + lo16;
      #pragma unroll
      for (int r = 0; r < 8; ++r)
        lds_th[(r + hi * 8) * SDIM + col] = f32_to_bf16(tanhf(h[t4].f[r]));
    }
    __syncthreads();

    // ---- phase B: h_prior = (1-1/tau)*h + (1/tau)*(th @ w_r^T + b_r)
    #pragma unroll
    for (int t4 = 0; t4 < 4; ++t4) {
      const int n = nbase + t4 * 16;
      v8f acc = {};
      #pragma unroll
      for (int kb = 0; kb < SDIM; kb += 32) {
        FragAB a, b;
        const int ka = kb + hi * 16;
        a.q[0] = *(const uint4*)(&lds_th[lo16 * SDIM + ka]);
        a.q[1] = *(const uint4*)(&lds_th[lo16 * SDIM + ka + 8]);
        const uint4* bp =
            (const uint4*)(&wr_b[(size_t)(n + lo16) * SDIM + kb + hi * 16]);
        b.q[0] = bp[0];
        b.q[1] = bp[1];
        acc = __builtin_amdgcn_wmma_f32_16x16x32_bf16(
            false, a.v, false, b.v, (short)0, acc, false, false);
      }
      FragC ac; ac.v = acc;
      const float br = lds_br[n + lo16];
      #pragma unroll
      for (int r = 0; r < 8; ++r) {
        float hp = (1.0f - TAU_INV) * h[t4].f[r] + TAU_INV * (ac.f[r] + br);
        h[t4].f[r] = hp;                       // h now holds h_prior
        float tv = tanhf(hp);
        tp[t4].f[r] = tv;
        lds_tp[(r + hi * 8) * SDIM + (n + lo16)] = f32_to_bf16(tv);
      }
    }
    __syncthreads();

    // ---- phase C (waves 0..3): x_pred = tanh(h_prior) @ w_o^T + b_o; error
    if (wave < 4) {
      const int n = wave * 16;                 // output-dim tile
      v8f acc = {};
      #pragma unroll
      for (int kb = 0; kb < SDIM; kb += 32) {
        FragAB a, b;
        const int ka = kb + hi * 16;
        a.q[0] = *(const uint4*)(&lds_tp[lo16 * SDIM + ka]);
        a.q[1] = *(const uint4*)(&lds_tp[lo16 * SDIM + ka + 8]);
        const uint4* bp =
            (const uint4*)(&wo_b[(size_t)(n + lo16) * SDIM + kb + hi * 16]);
        b.q[0] = bp[0];
        b.q[1] = bp[1];
        acc = __builtin_amdgcn_wmma_f32_16x16x32_bf16(
            false, a.v, false, b.v, (short)0, acc, false, false);
      }
      FragC e; e.v = acc;
      const int o  = n + lo16;
      const float bo = lds_bo[o];
      #pragma unroll
      for (int r = 0; r < 8; ++r) {
        const int m = r + hi * 8;
        const size_t idx = (size_t)t * (BATCH * ODIM) + (size_t)(m0 + m) * ODIM + o;
        float err = e.f[r] + bo - x[idx];
        out[idx] = err;
        lds_er[m * ODIM + o] = f32_to_bf16(err);
      }
    }
    __syncthreads();

    // ---- phase D: corr = err @ w_o (via w_o^T in LDS); h_post update
    #pragma unroll
    for (int t4 = 0; t4 < 4; ++t4) {
      const int n = nbase + t4 * 16;
      v8f acc = {};
      #pragma unroll
      for (int kb = 0; kb < ODIM; kb += 32) {
        FragAB a, b;
        const int ka = kb + hi * 16;
        a.q[0] = *(const uint4*)(&lds_er[lo16 * ODIM + ka]);
        a.q[1] = *(const uint4*)(&lds_er[lo16 * ODIM + ka + 8]);
        const int kB = kb + hi * 16;
        b.q[0] = *(const uint4*)(&lds_woT[(n + lo16) * ODIM + kB]);
        b.q[1] = *(const uint4*)(&lds_woT[(n + lo16) * ODIM + kB + 8]);
        acc = __builtin_amdgcn_wmma_f32_16x16x32_bf16(
            false, a.v, false, b.v, (short)0, acc, false, false);
      }
      FragC c; c.v = acc;
      #pragma unroll
      for (int r = 0; r < 8; ++r) {
        const float tv = tp[t4].f[r];
        h[t4].f[r] = h[t4].f[r] - ALPHA_X * (1.0f - tv * tv) * c.f[r];
      }
    }
    // No barrier needed here: next writes to lds_th/lds_er are separated from
    // this iteration's reads by the phase-A and phase-B barriers above.
  }
}

// ---------------------------------------------------------------------------
extern "C" void kernel_launch(void* const* d_in, const int* in_sizes, int n_in,
                              void* d_out, int out_size, void* d_ws, size_t ws_size,
                              hipStream_t stream) {
  const float* x      = (const float*)d_in[0];
  const float* h_init = (const float*)d_in[1];
  const float* w_o    = (const float*)d_in[2];
  const float* b_o    = (const float*)d_in[3];
  const float* w_r    = (const float*)d_in[4];
  const float* b_r    = (const float*)d_in[5];

  __bf16* wr_b  = (__bf16*)d_ws;                 // 512*512 bf16 = 512 KB
  __bf16* wo_b  = wr_b + (size_t)SDIM * SDIM;    // 64*512  bf16 =  64 KB
  __bf16* woT_b = wo_b + (size_t)ODIM * SDIM;    // 512*64  bf16 =  64 KB

  pc_rnn_prep_kernel<<<(SDIM * SDIM + 255) / 256, 256, 0, stream>>>(
      w_r, w_o, wr_b, wo_b, woT_b);

  pc_rnn_main_kernel<<<BATCH / 16, 256, 0, stream>>>(
      x, h_init, b_o, b_r, wr_b, wo_b, woT_b, (float*)d_out);
}
